// MultiQueryAttention_73495480369456
// MI455X (gfx1250) — compile-verified, matrix operand order checked by reading, not run on hardware
//
#include <hip/hip_runtime.h>

// MQA: B=1, S=2048, E=2048, H=16, D=128  (gfx1250 / wave32 / WMMA bf16)
#define S_LEN 2048
#define EMB   2048
#define NH    16
#define HD    128

typedef __attribute__((ext_vector_type(16))) __bf16 v16bf;
typedef __attribute__((ext_vector_type(8)))  float  v8f;

// Load one 16x32(A) or 32x16(B) bf16 WMMA fragment chunk for this lane.
// ISA layout: halves 0..7 = k0..k0+7, halves 8..15 = k0+16..k0+23, where the
// caller folds lane-dependent k0 = ((lane>>4)<<3) and m/n = (lane&15) into p.
__device__ __forceinline__ v16bf load_frag16(const __bf16* p) {
    union { v16bf v; uint4 q[2]; } u;
    u.q[0] = *reinterpret_cast<const uint4*>(p);
    u.q[1] = *reinterpret_cast<const uint4*>(p + 16);
    return u.v;
}

// ---------------------------------------------------------------- converts
__global__ __launch_bounds__(256)
void f32_to_bf16_kernel(const float* __restrict__ in, __bf16* __restrict__ out, int n) {
    int i = blockIdx.x * 256 + threadIdx.x;
    if (i < n) out[i] = (__bf16)in[i];
}

__global__ __launch_bounds__(256)
void v_transpose_kernel(const float* __restrict__ Vf, __bf16* __restrict__ Vt) {
    int i = blockIdx.x * 256 + threadIdx.x;     // over S*HD
    int s = i >> 7, d = i & 127;
    Vt[(size_t)d * S_LEN + s] = (__bf16)Vf[i];
}

// ---------------------------------------------------------------- GEMM
// C[M,N](f32) = A[M,K](bf16,row-major) x B[N,K](bf16,row-major)^T
// 128 threads = 4 waves (2x2); block tile 128x128; wave tile 64x64 (4x4 WMMA).
// B fragments register-resident per k-step, A streamed -> 16 WMMA / 8 frag loads.
__global__ __launch_bounds__(128)
void gemm_bf16_nt(const __bf16* __restrict__ A, const __bf16* __restrict__ B,
                  float* __restrict__ C, int M, int N, int K) {
    const int lane = threadIdx.x & 31;
    const int wave = threadIdx.x >> 5;
    const int wm = wave & 1, wn = wave >> 1;
    const int mbase = blockIdx.y * 128 + wm * 64;
    const int nbase = blockIdx.x * 128 + wn * 64;
    const int lr = lane & 15;
    const int lk = (lane >> 4) << 3;

    const __bf16* pa = A + (size_t)(mbase + lr) * K + lk;
    const __bf16* pb = B + (size_t)(nbase + lr) * K + lk;
    const size_t row16 = (size_t)16 * K;

    v8f c[4][4];
    #pragma unroll
    for (int i = 0; i < 4; ++i)
        #pragma unroll
        for (int j = 0; j < 4; ++j) c[i][j] = (v8f){};

    for (int k0 = 0; k0 < K; k0 += 32) {
        __builtin_prefetch(pa + k0 + 128, 0, 1);
        __builtin_prefetch(pb + k0 + 128, 0, 1);
        v16bf b[4];
        #pragma unroll
        for (int j = 0; j < 4; ++j) b[j] = load_frag16(pb + (size_t)j * row16 + k0);
        #pragma unroll
        for (int i = 0; i < 4; ++i) {
            v16bf a = load_frag16(pa + (size_t)i * row16 + k0);
            #pragma unroll
            for (int j = 0; j < 4; ++j)
                c[i][j] = __builtin_amdgcn_wmma_f32_16x16x32_bf16(
                              false, a, false, b[j], (short)0, c[i][j], false, false);
        }
    }
    const int ro = 8 * (lane >> 4);
    #pragma unroll
    for (int i = 0; i < 4; ++i) {
        #pragma unroll
        for (int r = 0; r < 8; ++r) {
            size_t m = (size_t)(mbase + i * 16 + ro + r);
            #pragma unroll
            for (int j = 0; j < 4; ++j)
                C[m * N + nbase + j * 16 + lr] = c[i][j][r];
        }
    }
}

// ------------------------------------------------- RMSNorm + double RoPE
// One wave per (h,s) row of 128. Qr = rope(rmsnorm(Q)) * 1/sqrt(D) (scale
// folded into Q operand of attention); Kr = rope(rope(rmsnorm(Q))) unscaled.
__global__ __launch_bounds__(128)
void qk_norm_rope(const float* __restrict__ Qf, const float* __restrict__ cosT,
                  const float* __restrict__ sinT, const float* __restrict__ qg,
                  __bf16* __restrict__ Qr, __bf16* __restrict__ Kr) {
    const int lane = threadIdx.x & 31;
    const int wave = threadIdx.x >> 5;
    const int row  = blockIdx.x * 4 + wave;       // 0 .. H*S-1
    const int h = row >> 11;                      // /2048
    const int s = row & (S_LEN - 1);

    const float* q = Qf + (size_t)s * EMB + h * HD;
    float a0 = q[lane], a1 = q[lane + 32], a2 = q[lane + 64], a3 = q[lane + 96];
    float ssq = a0 * a0 + a1 * a1 + a2 * a2 + a3 * a3;
    #pragma unroll
    for (int m = 16; m >= 1; m >>= 1) ssq += __shfl_xor(ssq, m, 32);
    float rn = rsqrtf(ssq * (1.0f / 128.0f) + 1e-6f);
    a0 *= rn * qg[lane];      a1 *= rn * qg[lane + 32];
    a2 *= rn * qg[lane + 64]; a3 *= rn * qg[lane + 96];

    const float* cp = cosT + (size_t)s * HD;
    const float* sp = sinT + (size_t)s * HD;
    float c0 = cp[lane], c1 = cp[lane + 32], c2 = cp[lane + 64], c3 = cp[lane + 96];
    float s0 = sp[lane], s1 = sp[lane + 32], s2 = sp[lane + 64], s3 = sp[lane + 96];

    // rope #1: y[d] = x[d]*cos[d] - x[d+64]*sin[d] ; y[d+64] = x[d+64]*cos + x[d]*sin
    float y0 = a0 * c0 - a2 * s0;
    float y1 = a1 * c1 - a3 * s1;
    float y2 = a2 * c2 + a0 * s2;
    float y3 = a3 * c3 + a1 * s3;
    // rope #2 (reference bug: K = rope(roped Q))
    float z0 = y0 * c0 - y2 * s0;
    float z1 = y1 * c1 - y3 * s1;
    float z2 = y2 * c2 + y0 * s2;
    float z3 = y3 * c3 + y1 * s3;

    __bf16* qo = Qr + ((size_t)h * S_LEN + s) * HD;
    __bf16* ko = Kr + ((size_t)h * S_LEN + s) * HD;
    const float scl = 0.08838834764831845f;       // 1/sqrt(128)
    qo[lane]      = (__bf16)(y0 * scl);
    qo[lane + 32] = (__bf16)(y1 * scl);
    qo[lane + 64] = (__bf16)(y2 * scl);
    qo[lane + 96] = (__bf16)(y3 * scl);
    ko[lane]      = (__bf16)z0;
    ko[lane + 32] = (__bf16)z1;
    ko[lane + 64] = (__bf16)z2;
    ko[lane + 96] = (__bf16)z3;
}

// ------------------------------------------------- flash attention (causal)
// One wave per (16-query block, head). Online softmax over 32-key blocks.
// QK^T: 2x(4 WMMA chained over d=128). P re-laid-out through LDS. P@V: 8 WMMA.
__global__ __launch_bounds__(32)
void mqa_flash(const __bf16* __restrict__ Qr, const __bf16* __restrict__ Kr,
               const __bf16* __restrict__ Vt, __bf16* __restrict__ AO) {
    __shared__ __attribute__((aligned(16))) __bf16 P[16 * 32];
    const int lane = threadIdx.x;
    const int h  = blockIdx.y;
    const int qb = blockIdx.x * 16;
    const int ln = lane & 15;
    const int hi = lane >> 4;
    const int lk = hi << 3;

    const __bf16* Qh = Qr + (size_t)h * S_LEN * HD;
    const __bf16* Kh = Kr + (size_t)h * S_LEN * HD;

    v16bf q[4];
    #pragma unroll
    for (int kk = 0; kk < 4; ++kk)
        q[kk] = load_frag16(Qh + (size_t)(qb + ln) * HD + kk * 32 + lk);

    v8f o[8];
    #pragma unroll
    for (int t = 0; t < 8; ++t) o[t] = (v8f){};
    float mrow[8], lrow[8];
    #pragma unroll
    for (int r = 0; r < 8; ++r) { mrow[r] = -1e30f; lrow[r] = 0.0f; }

    const int kend = (((qb + 15) >> 5) + 1) << 5;     // key blocks covering keys <= qb+15
    for (int jb = 0; jb < kend; jb += 32) {
        // ---- scores: two 16x16 tiles (keys jb..jb+15, jb+16..jb+31)
        v8f scv[2];
        #pragma unroll
        for (int t = 0; t < 2; ++t) {
            v8f acc = {};
            const __bf16* kp = Kh + (size_t)(jb + t * 16 + ln) * HD + lk;
            #pragma unroll
            for (int kk = 0; kk < 4; ++kk) {
                v16bf bf = load_frag16(kp + kk * 32);
                acc = __builtin_amdgcn_wmma_f32_16x16x32_bf16(false, q[kk], false, bf, (short)0, acc, false, false);
            }
            scv[t] = acc;
        }
        // ---- causal mask + online softmax row stats (row m = qb + r + 8*hi)
        float alpha[8];
        #pragma unroll
        for (int r = 0; r < 8; ++r) {
            int qrow = qb + r + 8 * hi;
            float s0v = (jb + ln      > qrow) ? -1e30f : scv[0][r];
            float s1v = (jb + 16 + ln > qrow) ? -1e30f : scv[1][r];
            float bm = fmaxf(s0v, s1v);
            #pragma unroll
            for (int msk = 8; msk >= 1; msk >>= 1) bm = fmaxf(bm, __shfl_xor(bm, msk, 32));
            float mnew = fmaxf(mrow[r], bm);
            float p0 = __expf(s0v - mnew);
            float p1 = __expf(s1v - mnew);
            float rs = p0 + p1;
            #pragma unroll
            for (int msk = 8; msk >= 1; msk >>= 1) rs += __shfl_xor(rs, msk, 32);
            alpha[r] = __expf(mrow[r] - mnew);
            lrow[r]  = lrow[r] * alpha[r] + rs;
            mrow[r]  = mnew;
            P[(r + 8 * hi) * 32 + ln]      = (__bf16)p0;
            P[(r + 8 * hi) * 32 + 16 + ln] = (__bf16)p1;
        }
        __syncthreads();
        // ---- rescale accumulators
        #pragma unroll
        for (int t = 0; t < 8; ++t) {
            #pragma unroll
            for (int r = 0; r < 8; ++r) o[t][r] = o[t][r] * alpha[r];
        }
        // ---- P @ V_block : A = P (16x32) from LDS, B = Vt rows (contiguous keys)
        v16bf pf = load_frag16(&P[ln * 32 + lk]);
        #pragma unroll
        for (int t = 0; t < 8; ++t) {
            v16bf vf = load_frag16(Vt + (size_t)(t * 16 + ln) * S_LEN + jb + lk);
            o[t] = __builtin_amdgcn_wmma_f32_16x16x32_bf16(false, pf, false, vf, (short)0, o[t], false, false);
        }
        __syncthreads();
    }
    // ---- epilogue: normalize, store AO[s][h*128+d] as bf16 (feeds Wo GEMM)
    __bf16* op = AO + (size_t)qb * EMB + h * HD;
    #pragma unroll
    for (int r = 0; r < 8; ++r) {
        float inv = 1.0f / lrow[r];
        #pragma unroll
        for (int t = 0; t < 8; ++t)
            op[(size_t)(r + 8 * hi) * EMB + t * 16 + ln] = (__bf16)(o[t][r] * inv);
    }
}

// ---------------------------------------------------------------- launch
extern "C" void kernel_launch(void* const* d_in, const int* in_sizes, int n_in,
                              void* d_out, int out_size, void* d_ws, size_t ws_size,
                              hipStream_t stream) {
    (void)in_sizes; (void)n_in; (void)out_size; (void)ws_size;
    const float* x    = (const float*)d_in[0];
    const float* cosT = (const float*)d_in[1];
    const float* sinT = (const float*)d_in[2];
    const float* Wq   = (const float*)d_in[3];
    /* d_in[4] = Wk — dead code in reference (K overwritten by rope(Q)) */
    const float* Wv   = (const float*)d_in[5];
    const float* Wo   = (const float*)d_in[6];
    const float* qg   = (const float*)d_in[7];
    /* d_in[8] = k_gamma — K rms-norm result is discarded */

    char* ws = (char*)d_ws;
    size_t off = 0;
    auto take = [&](size_t bytes) -> char* {
        char* p = ws + off;
        off += (bytes + 255) & ~(size_t)255;
        return p;
    };
    __bf16* xb  = (__bf16*)take((size_t)S_LEN * EMB * 2);
    __bf16* Wqb = (__bf16*)take((size_t)EMB * EMB * 2);
    __bf16* Wvb = (__bf16*)take((size_t)HD * EMB * 2);
    __bf16* Wob = (__bf16*)take((size_t)EMB * EMB * 2);
    float*  Qf  = (float*) take((size_t)S_LEN * EMB * 4);
    float*  Vf  = (float*) take((size_t)S_LEN * HD * 4);
    __bf16* Vt  = (__bf16*)take((size_t)HD * S_LEN * 2);
    __bf16* Qr  = (__bf16*)take((size_t)NH * S_LEN * HD * 2);
    __bf16* Kr  = (__bf16*)take((size_t)NH * S_LEN * HD * 2);
    __bf16* AOb = (__bf16*)Qf;   // Qf is dead after qk_norm_rope; reuse region

    // 1) f32 -> bf16 conversions
    f32_to_bf16_kernel<<<(S_LEN * EMB + 255) / 256, 256, 0, stream>>>(x,  xb,  S_LEN * EMB);
    f32_to_bf16_kernel<<<(EMB * EMB + 255) / 256,   256, 0, stream>>>(Wq, Wqb, EMB * EMB);
    f32_to_bf16_kernel<<<(HD * EMB + 255) / 256,    256, 0, stream>>>(Wv, Wvb, HD * EMB);
    f32_to_bf16_kernel<<<(EMB * EMB + 255) / 256,   256, 0, stream>>>(Wo, Wob, EMB * EMB);

    // 2) Q = x @ Wq^T  (f32 out for RMSNorm)
    gemm_bf16_nt<<<dim3(EMB / 128, S_LEN / 128), 128, 0, stream>>>(xb, Wqb, Qf, S_LEN, EMB, EMB);
    // 3) V = x @ Wv^T
    gemm_bf16_nt<<<dim3(HD / 128, S_LEN / 128), 128, 0, stream>>>(xb, Wvb, Vf, S_LEN, HD, EMB);
    // 4) Vt[d][s] = bf16(V[s][d])
    v_transpose_kernel<<<(S_LEN * HD) / 256, 256, 0, stream>>>(Vf, Vt);
    // 5) RMSNorm + RoPE -> Qr (scaled), Kr (double-roped)
    qk_norm_rope<<<(NH * S_LEN) / 4, 128, 0, stream>>>(Qf, cosT, sinT, qg, Qr, Kr);
    // 6) causal flash attention -> AO[s][h*128+d] bf16
    mqa_flash<<<dim3(S_LEN / 16, NH), 32, 0, stream>>>(Qr, Kr, Vt, AOb);
    // 7) out = AO @ Wo^T  (f32 into d_out)
    gemm_bf16_nt<<<dim3(EMB / 128, S_LEN / 128), 128, 0, stream>>>(AOb, Wob, (float*)d_out, S_LEN, EMB, EMB);
}
